// SlidingWindowAttention_25451976196662
// MI455X (gfx1250) — matrix operand order repaired and checked by reading, not compiled, not run on hardware
//
#include <hip/hip_runtime.h>
#include <hip/hip_bf16.h>

// ---------------------------------------------------------------------------
// Dilated sliding-window attention for MI455X (gfx1250), bf16 WMMA pipeline.
//   x[3072,1024] --(QKV bf16 GEMM)--> Q,K (row-major per head/phase), V^T
//   per (head, phase): flash attention over 768-token phase sequence, win=128
//   ctx --(output GEMM + bias)--> out f32 [3072,1024]
// ---------------------------------------------------------------------------

typedef __bf16 v16bf __attribute__((ext_vector_type(16)));
typedef __bf16 v8bf  __attribute__((ext_vector_type(8)));
typedef float  v8f   __attribute__((ext_vector_type(8)));

#define SEQ       3072
#define DMODEL    1024
#define NHEADS    16
#define DHEAD     64
#define WIN       128
#define DIL       4
#define PHASE_LEN (SEQ / DIL)      /* 768 tokens per (head,phase) stream */
#define TT        (PHASE_LEN / 16) /* 48 query tiles per stream          */

// round-to-nearest-even f32 -> bf16
__device__ __forceinline__ __bf16 f2bf(float f) {
  unsigned u = __builtin_bit_cast(unsigned, f);
  u += 0x7FFFu + ((u >> 16) & 1u);
  unsigned short h = (unsigned short)(u >> 16);
  return __builtin_bit_cast(__bf16, h);
}

__device__ __forceinline__ v8f wmma_bf16(v16bf a, v16bf b, v8f c) {
  // D(16x16,f32) = A(16x32,bf16) x B(32x16,bf16) + C
  return __builtin_amdgcn_wmma_f32_16x16x32_bf16(false, a, false, b,
                                                 (short)0, c, false, false);
}

// A-fragment (16x32 bf16): lane-half h holds K = kbase+16h+[0..15]
// in elems 0..15 (contiguous, matching B); row = lane&15.
__device__ __forceinline__ v16bf load_a_frag(const __bf16* row, int kbase, int half) {
  v8bf lo = *(const v8bf*)(row + kbase + 16 * half);
  v8bf hi = *(const v8bf*)(row + kbase + 16 * half + 8);
  v16bf a;
#pragma unroll
  for (int i = 0; i < 8; i++) { a[i] = lo[i]; a[i + 8] = hi[i]; }
  return a;
}

// ---------------------------------------------------------------------------
__global__ void cvt_bf16(const float* __restrict__ src, __bf16* __restrict__ dst, int n) {
  int i = blockIdx.x * blockDim.x + threadIdx.x;
  int stride = gridDim.x * blockDim.x;
  for (; i < n; i += stride) dst[i] = f2bf(src[i]);
}

// ---------------------------------------------------------------------------
// Fused QKV projection. wave -> 16(token) x 64(out-col) strip of one of Q/K/V.
// M tiles iterate *within a dilation phase* so stores land contiguously in the
// per-(head,phase) layouts used by the attention kernel.
//   Q,K : [h][p][t][64] row-major (Q pre-scaled by exp(-beta)/8)
//   V   : [h][p][64][t] transposed (so PV B-fragments are contiguous)
__global__ void __launch_bounds__(128)
qkv_gemm(const __bf16* __restrict__ xb,
         const __bf16* __restrict__ Wqb, const __bf16* __restrict__ Wkb,
         const __bf16* __restrict__ Wvb,
         const float* __restrict__ bq, const float* __restrict__ bk,
         const float* __restrict__ bv, const float* __restrict__ beta,
         __bf16* __restrict__ Qb, __bf16* __restrict__ Kb, __bf16* __restrict__ Vtb)
{
  const int lane = threadIdx.x & 31;
  const int wid  = threadIdx.x >> 5;
  const int gw   = blockIdx.x * 4 + wid;          // 9216 waves total
  const int nstrip = gw & 15;                     // 16 N strips of 64
  int rem = gw >> 4;
  const int tt  = rem % TT;  rem /= TT;           // 48 token tiles / phase
  const int p   = rem & 3;                        // dilation phase
  const int mat = rem >> 2;                       // 0=Q 1=K 2=V

  const __bf16* W   = (mat == 0) ? Wqb : ((mat == 1) ? Wkb : Wvb);
  const float* bias = (mat == 0) ? bq  : ((mat == 1) ? bk  : bv);

  const int half = lane >> 4;
  const int ln   = lane & 15;
  const int t_base = tt * 16;
  const int arow = 4 * (t_base + ln) + p;         // token index for this A row
  const __bf16* arow_ptr = xb + (size_t)arow * DMODEL;
  const int n0 = nstrip * 64;

  v8f acc[4] = {};
  for (int kk = 0; kk < DMODEL; kk += 32) {
    v16bf a = load_a_frag(arow_ptr, kk, half);
#pragma unroll
    for (int j = 0; j < 4; j++) {
      const __bf16* bp = W + (size_t)(n0 + j * 16 + ln) * DMODEL + kk + 16 * half;
      v16bf b = *(const v16bf*)bp;                // 32B contiguous per lane
      acc[j] = wmma_bf16(a, b, acc[j]);
      if (kk + 32 < DMODEL) __builtin_prefetch(bp + 32, 0, 3);
    }
  }

#pragma unroll
  for (int j = 0; j < 4; j++) {
    const int o = n0 + j * 16 + ln;               // output column (per lane)
    const int h = o >> 6, d = o & 63;
    const float bi = bias[o];
    float scale = 1.0f;
    if (mat == 0) scale = 0.125f * __expf(-beta[h]);  // 1/(sqrt(64)*e^beta)
#pragma unroll
    for (int r = 0; r < 8; r++) {
      const int t = t_base + r + 8 * half;        // C layout: M = r + 8*half
      const float v = (acc[j][r] + bi) * scale;
      const __bf16 bvv = f2bf(v);
      if (mat == 2)
        Vtb[((size_t)(h * 4 + p) * 64 + d) * PHASE_LEN + t] = bvv;
      else {
        __bf16* dst = (mat == 0) ? Qb : Kb;
        dst[((size_t)(h * 4 + p) * PHASE_LEN + t) * 64 + d] = bvv;
      }
    }
  }
}

// ---------------------------------------------------------------------------
// Flash attention per (head, phase, 16-query tile). Window 128 in the phase
// domain => key tiles qt-8 .. qt, processed 32 keys (2 tiles) per step.
__global__ void __launch_bounds__(128)
attn_kernel(const __bf16* __restrict__ Qb, const __bf16* __restrict__ Kb,
            const __bf16* __restrict__ Vtb, __bf16* __restrict__ ctxb)
{
  __shared__ __bf16 ldsP[4][16 * 32];             // per-wave P tile (1 KB)

  const int lane = threadIdx.x & 31;
  const int wid  = threadIdx.x >> 5;
  const int gw   = blockIdx.x * 4 + wid;          // 3072 waves total
  const int qt   = gw % TT;
  const int p    = (gw / TT) & 3;
  const int h    = gw / (TT * 4);
  const int half = lane >> 4;
  const int ln   = lane & 15;

  const __bf16* Qhp = Qb  + (size_t)(h * 4 + p) * PHASE_LEN * 64;
  const __bf16* Khp = Kb  + (size_t)(h * 4 + p) * PHASE_LEN * 64;
  const __bf16* Vhp = Vtb + (size_t)(h * 4 + p) * 64 * PHASE_LEN;

  // Q A-fragments, d-chunks [0,32) and [32,64)
  const __bf16* qrow = Qhp + (size_t)(qt * 16 + ln) * 64;
  const v16bf aq0 = load_a_frag(qrow, 0, half);
  const v16bf aq1 = load_a_frag(qrow, 32, half);

  float m[8], lsum[8];
  v8f acc[4] = {};
#pragma unroll
  for (int r = 0; r < 8; r++) { m[r] = -1e30f; lsum[r] = 0.f; }

  const int t_row0 = qt * 16 + 8 * half;

  for (int kt = qt - 8; kt <= qt; kt += 2) {
    if (kt + 1 < 0) continue;                     // both tiles before t=0
    v8f s[2];
#pragma unroll
    for (int c2 = 0; c2 < 2; c2++) {              // two 16-key tiles
      const int tile  = kt + c2;
      const bool valid = (tile >= 0) && (tile < TT);
      const int uld   = (valid ? tile : 0) * 16;  // clamped load base
      const __bf16* krow = Khp + (size_t)(uld + ln) * 64;
      v16bf bk0 = *(const v16bf*)(krow + 16 * half);        // d 0..31
      v16bf bk1 = *(const v16bf*)(krow + 32 + 16 * half);   // d 32..63
      v8f sc = {};
      sc = wmma_bf16(aq0, bk0, sc);
      sc = wmma_bf16(aq1, bk1, sc);
      const int u = tile * 16 + ln;               // key index (per lane)
#pragma unroll
      for (int r = 0; r < 8; r++) {
        const int t = t_row0 + r;
        const bool ok = valid && (u <= t) && (u >= t - (WIN - 1));
        s[c2][r] = ok ? sc[r] : -1e30f;
      }
    }

    // online softmax over these 32 keys (row reductions across 16-lane groups)
    float alpha[8];
#pragma unroll
    for (int r = 0; r < 8; r++) {
      float v = fmaxf(s[0][r], s[1][r]);
      v = fmaxf(v, __shfl_xor(v, 1));
      v = fmaxf(v, __shfl_xor(v, 2));
      v = fmaxf(v, __shfl_xor(v, 4));
      v = fmaxf(v, __shfl_xor(v, 8));
      const float mn = fmaxf(m[r], v);
      alpha[r] = __expf(m[r] - mn);
      m[r] = mn;
      const float p0 = __expf(s[0][r] - mn);
      const float p1 = __expf(s[1][r] - mn);
      const int row = r + 8 * half;               // C layout -> LDS [q][32]
      ldsP[wid][row * 32 + ln]      = f2bf(p0);
      ldsP[wid][row * 32 + 16 + ln] = f2bf(p1);
      float rs = p0 + p1;
      rs += __shfl_xor(rs, 1);
      rs += __shfl_xor(rs, 2);
      rs += __shfl_xor(rs, 4);
      rs += __shfl_xor(rs, 8);
      lsum[r] = lsum[r] * alpha[r] + rs;
    }
#pragma unroll
    for (int j = 0; j < 4; j++)
#pragma unroll
      for (int r = 0; r < 8; r++) acc[j][r] *= alpha[r];

    // P as A-fragment (16 queries x 32 keys) via LDS transpose round-trip
    const v16bf ap = load_a_frag(&ldsP[wid][(size_t)ln * 32], 0, half);

    // PV: B = V^T chunk [32 keys x 16 dims]; lane-half selects the key tile,
    // so clamp the base per half (clamped halves carry P==0 anyway).
#pragma unroll
    for (int j = 0; j < 4; j++) {
      int th = kt + half;
      th = (th < 0) ? 0 : ((th >= TT) ? TT - 1 : th);
      const __bf16* vrow = Vhp + (size_t)(j * 16 + ln) * PHASE_LEN + th * 16;
      v16bf bv = *(const v16bf*)vrow;             // 32B contiguous per lane
      acc[j] = wmma_bf16(ap, bv, acc[j]);
    }
  }

  // normalize and scatter to ctx[s][d_model] (token s = 4t + p)
  float inv[8];
#pragma unroll
  for (int r = 0; r < 8; r++) inv[r] = 1.0f / lsum[r];
#pragma unroll
  for (int j = 0; j < 4; j++) {
#pragma unroll
    for (int r = 0; r < 8; r++) {
      const int t = qt * 16 + r + 8 * half;
      const int stok = 4 * t + p;
      const int col = h * 64 + j * 16 + ln;
      ctxb[(size_t)stok * DMODEL + col] = f2bf(acc[j][r] * inv[r]);
    }
  }
}

// ---------------------------------------------------------------------------
// Output projection: out = ctx @ Wo^T + bo (f32 result)
__global__ void __launch_bounds__(128)
out_gemm(const __bf16* __restrict__ ctxb, const __bf16* __restrict__ Wob,
         const float* __restrict__ bo, float* __restrict__ out)
{
  const int lane = threadIdx.x & 31;
  const int wid  = threadIdx.x >> 5;
  const int gw   = blockIdx.x * 4 + wid;          // 3072 waves total
  const int nstrip = gw & 15;
  const int st     = gw >> 4;                     // 0..191 token tiles
  const int half = lane >> 4;
  const int ln   = lane & 15;

  const __bf16* arow = ctxb + (size_t)(st * 16 + ln) * DMODEL;
  const int n0 = nstrip * 64;

  v8f acc[4] = {};
  for (int kk = 0; kk < DMODEL; kk += 32) {
    v16bf a = load_a_frag(arow, kk, half);
#pragma unroll
    for (int j = 0; j < 4; j++) {
      const __bf16* bp = Wob + (size_t)(n0 + j * 16 + ln) * DMODEL + kk + 16 * half;
      v16bf b = *(const v16bf*)bp;
      acc[j] = wmma_bf16(a, b, acc[j]);
      if (kk + 32 < DMODEL) __builtin_prefetch(bp + 32, 0, 3);
    }
  }
#pragma unroll
  for (int j = 0; j < 4; j++) {
    const int o = n0 + j * 16 + ln;
    const float bi = bo[o];
#pragma unroll
    for (int r = 0; r < 8; r++) {
      const int s = st * 16 + r + 8 * half;
      out[(size_t)s * DMODEL + o] = acc[j][r] + bi;
    }
  }
}

// ---------------------------------------------------------------------------
extern "C" void kernel_launch(void* const* d_in, const int* in_sizes, int n_in,
                              void* d_out, int out_size, void* d_ws, size_t ws_size,
                              hipStream_t stream) {
  const float* x    = (const float*)d_in[0];
  const float* beta = (const float*)d_in[1];
  const float* Wq   = (const float*)d_in[2];
  const float* bq   = (const float*)d_in[3];
  const float* Wk   = (const float*)d_in[4];
  const float* bk   = (const float*)d_in[5];
  const float* Wv   = (const float*)d_in[6];
  const float* bv   = (const float*)d_in[7];
  const float* Wo   = (const float*)d_in[8];
  const float* bo   = (const float*)d_in[9];
  float* out = (float*)d_out;

  const size_t NX = (size_t)SEQ * DMODEL;     // 3,145,728
  const size_t NW = (size_t)DMODEL * DMODEL;  // 1,048,576

  char* ws = (char*)d_ws;
  __bf16* xb   = (__bf16*)ws; ws += NX * 2;
  __bf16* Wqb  = (__bf16*)ws; ws += NW * 2;
  __bf16* Wkb  = (__bf16*)ws; ws += NW * 2;
  __bf16* Wvb  = (__bf16*)ws; ws += NW * 2;
  __bf16* Wob  = (__bf16*)ws; ws += NW * 2;
  __bf16* Qb   = (__bf16*)ws; ws += NX * 2;
  __bf16* Kb   = (__bf16*)ws; ws += NX * 2;
  __bf16* Vtb  = (__bf16*)ws; ws += NX * 2;
  __bf16* ctxb = (__bf16*)ws; ws += NX * 2;   // total ~38 MB

  cvt_bf16<<<512, 256, 0, stream>>>(x,  xb,  (int)NX);
  cvt_bf16<<<512, 256, 0, stream>>>(Wq, Wqb, (int)NW);
  cvt_bf16<<<512, 256, 0, stream>>>(Wk, Wkb, (int)NW);
  cvt_bf16<<<512, 256, 0, stream>>>(Wv, Wvb, (int)NW);
  cvt_bf16<<<512, 256, 0, stream>>>(Wo, Wob, (int)NW);

  // 3 mats * 4 phases * 48 tiles * 16 strips = 9216 waves / 4 per block
  qkv_gemm<<<2304, 128, 0, stream>>>(xb, Wqb, Wkb, Wvb, bq, bk, bv, beta,
                                     Qb, Kb, Vtb);
  // 16 heads * 4 phases * 48 query tiles = 3072 waves / 4 per block
  attn_kernel<<<768, 128, 0, stream>>>(Qb, Kb, Vtb, ctxb);
  // 192 token tiles * 16 strips = 3072 waves / 4 per block
  out_gemm<<<768, 128, 0, stream>>>(ctxb, Wob, bo, out);
}
